// Up_86535001080407
// MI455X (gfx1250) — compile-verified, matrix operand order checked by reading, not run on hardware
//
#include <hip/hip_runtime.h>

// ---------------------------------------------------------------------------
// CDNA5 (gfx1250) implementation: wave32 + v_wmma_f32_16x16x32_bf16 for all
// GEMM-able stages; conv input tiles staged with GLOBAL_LOAD_ASYNC_TO_LDS_B128
// (ASYNCcnt); BN via single-pass sum/sumsq (ds_add_f32 + global atomics).
// ---------------------------------------------------------------------------

typedef __attribute__((ext_vector_type(16))) __bf16 v16bf;
typedef __attribute__((ext_vector_type(8)))  float  v8f;

#define EPSBN 1e-5f
#define BS   2
#define T_   8
#define H_   128
#define W_   128
#define HW   (H_*W_)        // 16384
#define DEC  64
#define ENC  32
#define NH   4
#define CCAT 96             // ENC + DEC
#define NPIX 262144         // BS*T_*HW (per-channel BN count)

// ---- workspace layout (bytes) ----------------------------------------------
static constexpr size_t WS_STATS  = 0;                       // 512 f32 (3 stages x 64)
static constexpr size_t WS_AFFINE = 2048;                    // 3 x 64 f32
static constexpr size_t WS_PK_UP  = 4096;                    // 16384 bf16
static constexpr size_t WS_PK_MLP = WS_PK_UP  + 32768;       // 4096 bf16
static constexpr size_t WS_PK_C1  = WS_PK_MLP + 8192;        // 27648 bf16
static constexpr size_t WS_PK_C2  = WS_PK_C1  + 55296;       // 9216 bf16
static constexpr size_t WS_HMLP   = 131072;                  // 8388608 f32
static constexpr size_t WS_CAT    = WS_HMLP + 33554432ull;   // 25165824 bf16
static constexpr size_t WS_Y1     = WS_CAT  + 50331648ull;   // 8388608 f32
static constexpr size_t WS_CAT2   = WS_Y1   + 33554432ull;   // 8388608 bf16
static constexpr size_t WS_Y2     = WS_CAT2 + 16777216ull;   // 8388608 f32

__device__ __forceinline__ v8f wmma_bf16(v16bf a, v16bf b, v8f c) {
  // D = A(16x32 bf16) * B(32x16 bf16) + C(16x16 f32)
  return __builtin_amdgcn_wmma_f32_16x16x32_bf16(false, a, false, b,
                                                 (short)0, c, false, false);
}

__device__ __forceinline__ int iclamp(int v, int lo, int hi) {
  return v < lo ? lo : (v > hi ? hi : v);
}

// Async global->LDS copy of 16 bytes (CDNA5 GLOBAL_LOAD_ASYNC_TO_LDS_B128).
// ldsoff: LDS byte address (low 32 bits of the flat shared address);
// gaddr : 64-bit global byte address. Tracked by ASYNCcnt.
__device__ __forceinline__ void async_copy_b128(unsigned ldsoff, const void* gaddr) {
  asm volatile("global_load_async_to_lds_b128 %0, %1, off"
               :: "v"(ldsoff), "v"((unsigned long long)(uintptr_t)gaddr)
               : "memory");
}
__device__ __forceinline__ void wait_asynccnt0() {
  asm volatile("s_wait_asynccnt 0x0" ::: "memory");
}

// ---------------------------------------------------------------------------
// K0: repack all weights to bf16 in WMMA B-operand lane layout:
//   pack[(tile*NKS+ks)*512 + lane*16 + j]  holds  B[k][n] with
//   n = tile*16 + (lane&15),  k = ks*32 + 8*(lane>=16) + (j<8 ? j : j+8)
// Also zero the BN stats region.
// ---------------------------------------------------------------------------
__global__ void k_pack(const float* __restrict__ Wup, const float* __restrict__ Wmlp,
                       const float* __restrict__ Wc1, const float* __restrict__ Wc2,
                       __bf16* pk_up, __bf16* pk_mlp, __bf16* pk_c1, __bf16* pk_c2,
                       float* stats) {
  int tid = blockIdx.x * blockDim.x + threadIdx.x;
  if (tid < 512) stats[tid] = 0.0f;

  int e = tid;
  if (e < 16384) {                                   // W_up: 16 nt x 2 ks
    int tk = e >> 9, lane = (e >> 4) & 31, j = e & 15;
    int nt = tk >> 1, ks = tk & 1;
    int n = nt * 16 + (lane & 15);
    int k = ks * 32 + ((lane >> 4) << 3) + (j < 8 ? j : j + 8);   // k = input c
    int o = n >> 2, ky = (n >> 1) & 1, kx = n & 1;
    pk_up[e] = (__bf16)Wup[(k * 64 + o) * 4 + ky * 2 + kx];
    return;
  }
  e -= 16384;
  if (e < 4096) {                                    // W_mlp: 2 nt x 4 ks
    int tk = e >> 9, lane = (e >> 4) & 31, j = e & 15;
    int nt = tk >> 2, ks = tk & 3;
    int n = nt * 16 + (lane & 15);
    int k = ks * 32 + ((lane >> 4) << 3) + (j < 8 ? j : j + 8);   // k = c*4+n_head
    pk_mlp[e] = (__bf16)Wmlp[k * 32 + n];
    return;
  }
  e -= 4096;
  if (e < 27648) {                                   // W_c1: 2 nt x 27 ks, K=864
    int tk = e >> 9, lane = (e >> 4) & 31, j = e & 15;
    int nt = tk / 27, ks = tk % 27;
    int n = nt * 16 + (lane & 15);
    int k = ks * 32 + ((lane >> 4) << 3) + (j < 8 ? j : j + 8);
    int ic = k / 9, r9 = k % 9;
    pk_c1[e] = (__bf16)Wc1[(n * CCAT + ic) * 9 + r9];
    return;
  }
  e -= 27648;
  if (e < 9216) {                                    // W_c2: 2 nt x 9 ks, K=288
    int tk = e >> 9, lane = (e >> 4) & 31, j = e & 15;
    int nt = tk / 9, ks = tk % 9;
    int n = nt * 16 + (lane & 15);
    int k = ks * 32 + ((lane >> 4) << 3) + (j < 8 ? j : j + 8);
    int ic = k / 9, r9 = k % 9;
    pk_c2[e] = (__bf16)Wc2[(n * ENC + ic) * 9 + r9];
  }
}

// ---------------------------------------------------------------------------
// K1: 2x learned upsample of x1.  GEMM per (b,t): [4096 pos x 64c] x [64 x 256].
// Writes bf16 into concat channels 32..95.
// ---------------------------------------------------------------------------
__global__ void k_upsample(const float* __restrict__ x1, const float* __restrict__ b_up,
                           const __bf16* __restrict__ pk, __bf16* __restrict__ cat) {
  int bt = blockIdx.y, b = bt >> 3, t = bt & 7;
  int wave = threadIdx.x >> 5, lane = threadIdx.x & 31;
  int mbase = blockIdx.x * 128 + wave * 16;        // coarse pixel tile
  int mrow  = mbase + (lane & 15);
  int kb8   = (lane >> 4) << 3;

  v16bf a0, a1;                                     // K = 0..31, 32..63 (= input ch)
#pragma unroll
  for (int j = 0; j < 16; ++j) {
    int kk = kb8 + (j < 8 ? j : j + 8);
    a0[j] = (__bf16)x1[(((size_t)b * DEC + kk)      * T_ + t) * 4096 + mrow];
    a1[j] = (__bf16)x1[(((size_t)b * DEC + 32 + kk) * T_ + t) * 4096 + mrow];
  }
  for (int nt = 0; nt < 16; ++nt) {
    v8f acc = {};
    acc = wmma_bf16(a0, *(const v16bf*)(pk + (size_t)((nt * 2 + 0) * 32 + lane) * 16), acc);
    acc = wmma_bf16(a1, *(const v16bf*)(pk + (size_t)((nt * 2 + 1) * 32 + lane) * 16), acc);
    int n = nt * 16 + (lane & 15);
    int o = n >> 2, ky = (n >> 1) & 1, kx = n & 1;
    float bias = b_up[o];
#pragma unroll
    for (int r = 0; r < 8; ++r) {
      int m = mbase + r + kb8;
      int h = m >> 6, w = m & 63;
      size_t oi = (((size_t)b * CCAT + 32 + o) * T_ + t) * HW
                + (size_t)(2 * h + ky) * W_ + (2 * w + kx);
      cat[oi] = (__bf16)(acc[r] + bias);
    }
  }
}

// ---------------------------------------------------------------------------
// K2: bilinear attention ctx + MLP projection (K=128 -> N=32), pre-BN f32 out
// plus per-channel sum/sumsq stats.
// ---------------------------------------------------------------------------
__global__ void k_attn_mlp(const float* __restrict__ x2, const float* __restrict__ attn,
                           const float* __restrict__ b_mlp, const __bf16* __restrict__ pk,
                           float* __restrict__ hmlp, float* __restrict__ stats) {
  __shared__ float G[NH * 16 * 16 * T_];           // coarse attn grid (n,i,j,s): 32 KB
  __shared__ float bsum[ENC], bsq[ENC];
  int tid = threadIdx.x;
  int bt = blockIdx.y, b = bt >> 3, t = bt & 7;

  for (int f = tid; f < NH * 16 * 16 * T_; f += 256) {
    int s = f & 7, j = (f >> 3) & 15, i = (f >> 7) & 15, n = f >> 11;
    G[f] = attn[(((((size_t)n * BS + b) * 16 + i) * 16 + j) * T_ + t) * T_ + s];
  }
  if (tid < ENC) { bsum[tid] = 0.0f; bsq[tid] = 0.0f; }
  __syncthreads();

  int wave = tid >> 5, lane = tid & 31;
  int mbase = blockIdx.x * 128 + wave * 16;
  int p = mbase + (lane & 15);
  int x = p >> 7, y = p & 127;
  int kb8 = (lane >> 4) << 3;

  // half-pixel bilinear coords, 16 -> 128 per axis
  float sx = (x + 0.5f) * 0.125f - 0.5f;
  float sy = (y + 0.5f) * 0.125f - 0.5f;
  int ix = (int)floorf(sx), iy = (int)floorf(sy);
  float fx = sx - ix, fy = sy - iy;
  int ix0 = iclamp(ix, 0, 15),     iy0 = iclamp(iy, 0, 15);
  int ix1 = iclamp(ix + 1, 0, 15), iy1 = iclamp(iy + 1, 0, 15);

  float a_ns[NH * T_];
#pragma unroll
  for (int n = 0; n < NH; ++n)
#pragma unroll
    for (int s = 0; s < T_; ++s) {
      float v00 = G[((n * 16 + ix0) * 16 + iy0) * T_ + s];
      float v01 = G[((n * 16 + ix0) * 16 + iy1) * T_ + s];
      float v10 = G[((n * 16 + ix1) * 16 + iy0) * T_ + s];
      float v11 = G[((n * 16 + ix1) * 16 + iy1) * T_ + s];
      a_ns[n * T_ + s] = (1.0f - fx) * ((1.0f - fy) * v00 + fy * v01)
                       +          fx * ((1.0f - fy) * v10 + fy * v11);
    }

  v8f acc0 = {}, acc1 = {};
#pragma unroll
  for (int ks = 0; ks < 4; ++ks) {
    v16bf av;
#pragma unroll
    for (int g4 = 0; g4 < 4; ++g4) {
      int j0 = g4 * 4;
      int rel0 = (j0 < 8) ? j0 : j0 + 8;           // {0,4,16,20}
      int k0 = ks * 32 + kb8 + rel0;               // multiple of 4
      int c = k0 >> 2;                             // shared by the 4 elements
      float s0 = 0.f, s1 = 0.f, s2 = 0.f, s3 = 0.f;
#pragma unroll
      for (int s = 0; s < T_; ++s) {
        float xv = x2[(((size_t)b * ENC + c) * T_ + s) * HW + p];
        s0 += a_ns[0 * T_ + s] * xv;
        s1 += a_ns[1 * T_ + s] * xv;
        s2 += a_ns[2 * T_ + s] * xv;
        s3 += a_ns[3 * T_ + s] * xv;
      }
      av[j0 + 0] = (__bf16)s0; av[j0 + 1] = (__bf16)s1;
      av[j0 + 2] = (__bf16)s2; av[j0 + 3] = (__bf16)s3;
    }
    acc0 = wmma_bf16(av, *(const v16bf*)(pk + (size_t)((0 * 4 + ks) * 32 + lane) * 16), acc0);
    acc1 = wmma_bf16(av, *(const v16bf*)(pk + (size_t)((1 * 4 + ks) * 32 + lane) * 16), acc1);
  }

#pragma unroll
  for (int nt = 0; nt < 2; ++nt) {
    v8f acc = nt ? acc1 : acc0;
    int o = nt * 16 + (lane & 15);
    float bias = b_mlp[o];
    float lsum = 0.f, lsq = 0.f;
#pragma unroll
    for (int r = 0; r < 8; ++r) {
      int pd = mbase + r + kb8;
      float hv = acc[r] + bias;
      hmlp[(((size_t)b * ENC + o) * T_ + t) * HW + pd] = hv;
      lsum += hv; lsq += hv * hv;
    }
    atomicAdd(&bsum[o], lsum);
    atomicAdd(&bsq[o], lsq);
  }
  __syncthreads();
  if (tid < ENC) {
    atomicAdd(&stats[tid], bsum[tid]);
    atomicAdd(&stats[32 + tid], bsq[tid]);
  }
}

// ---------------------------------------------------------------------------
// BN stats -> per-channel affine (a, c):  y = a*x + c
// ---------------------------------------------------------------------------
__global__ void k_affine(const float* __restrict__ stats, const float* __restrict__ g,
                         const float* __restrict__ be, float* __restrict__ affine) {
  int c = threadIdx.x;
  if (c < ENC) {
    float mean = stats[c] * (1.0f / (float)NPIX);
    float var  = stats[32 + c] * (1.0f / (float)NPIX) - mean * mean;
    float a = g[c] * rsqrtf(var + EPSBN);
    affine[c] = a;
    affine[32 + c] = be[c] - mean * a;
  }
}

// ---------------------------------------------------------------------------
// Elementwise BN-apply + ReLU.  bf16 variant writes into a buffer with `outc`
// channels at channel offset `choff` (used to fill the concat buffer).
// ---------------------------------------------------------------------------
__global__ void k_bnrelu_bf16(const float* __restrict__ in, const float* __restrict__ affine,
                              __bf16* __restrict__ out, int outc, int choff) {
  size_t i = (size_t)blockIdx.x * 256 + threadIdx.x;    // over BS*ENC*T_*HW
  const size_t CH = (size_t)T_ * HW;
  int ch = (int)((i / CH) & (ENC - 1));
  int b  = (int)(i / (CH * ENC));
  float v = fmaf(affine[ch], in[i], affine[32 + ch]);
  v = fmaxf(v, 0.0f);
  size_t oi = ((size_t)b * outc + choff + ch) * CH + (i % CH);
  out[oi] = (__bf16)v;
}

__global__ void k_bnrelu_f32(const float* __restrict__ in, const float* __restrict__ affine,
                             float* __restrict__ out) {
  size_t i = (size_t)blockIdx.x * 256 + threadIdx.x;
  const size_t CH = (size_t)T_ * HW;
  int ch = (int)((i / CH) & (ENC - 1));
  float v = fmaf(affine[ch], in[i], affine[32 + ch]);
  out[i] = fmaxf(v, 0.0f);
}

// ---------------------------------------------------------------------------
// 1x3x3 conv as im2col WMMA.  One output row (128 px) per block, 8 waves x 16.
// Input halo tile (IC x 3 rows) staged in LDS via async-to-LDS b128 copies.
// Row stride 144 elems (288 B): valid data at cols 8..135 (16B-aligned chunks),
// zero halo at cols 7 and 136.  LDS col for (pixel y, tap dc) = 7 + y + dc.
// K = IC*9, NKS = K/32 k-steps; N = 32 output channels (2 n-tiles).
// ---------------------------------------------------------------------------
template<int IC, int NKS>
__global__ void k_conv3x3(const __bf16* __restrict__ in, const __bf16* __restrict__ pk,
                          const float* __restrict__ bias, float* __restrict__ out,
                          float* __restrict__ stats) {
  constexpr int RSTR = 144;                         // row stride (elems), 288 B
  __shared__ __attribute__((aligned(16))) __bf16 tile[IC * 3 * RSTR];
  __shared__ float bsum[ENC], bsq[ENC];
  int tid = threadIdx.x;
  int bt = blockIdx.y, b = bt >> 3, t = bt & 7;
  int xr = blockIdx.x;                              // output row

  if (tid < ENC) { bsum[tid] = 0.0f; bsq[tid] = 0.0f; }

  // 1) zero halo columns; zero full rows whose source x is out of range
  const int NROW = IC * 3;
  for (int rrow = tid; rrow < NROW; rrow += 256) {
    int dr = rrow % 3;
    int gx = xr - 1 + dr;
    tile[rrow * RSTR + 7]   = (__bf16)0.0f;
    tile[rrow * RSTR + 136] = (__bf16)0.0f;
    if (gx < 0 || gx >= H_) {
      for (int c = 8; c < 136; ++c) tile[rrow * RSTR + c] = (__bf16)0.0f;
    }
  }

  // 2) async global->LDS copies of the 128 valid bf16 per row (16 x b128)
  const int NCH = NROW * 16;
  for (int f = tid; f < NCH; f += 256) {
    int ch = f & 15, rrow = f >> 4;
    int dr = rrow % 3, ic = rrow / 3;
    int gx = xr - 1 + dr;
    if (gx >= 0 && gx < H_) {
      int gy = ch * 8;
      const __bf16* src = in + (((size_t)b * IC + ic) * T_ + t) * HW
                             + (size_t)gx * W_ + gy;
      unsigned ldsoff = (unsigned)(uintptr_t)&tile[rrow * RSTR + 8 + gy];
      async_copy_b128(ldsoff, src);
    }
  }
  wait_asynccnt0();
  __syncthreads();

  int wave = tid >> 5, lane = tid & 31;
  int ybase = wave * 16;
  int ycol  = ybase + (lane & 15);
  int kb8   = (lane >> 4) << 3;

  v8f acc0 = {}, acc1 = {};
  for (int ks = 0; ks < NKS; ++ks) {
    v16bf av;
#pragma unroll
    for (int j = 0; j < 16; ++j) {
      int k = ks * 32 + kb8 + (j < 8 ? j : j + 8);
      int ic = k / 9, r9 = k % 9;
      int dr = r9 / 3, dc = r9 % 3;
      av[j] = tile[(ic * 3 + dr) * RSTR + 7 + ycol + dc];
    }
    acc0 = wmma_bf16(av, *(const v16bf*)(pk + (size_t)((0 * NKS + ks) * 32 + lane) * 16), acc0);
    acc1 = wmma_bf16(av, *(const v16bf*)(pk + (size_t)((1 * NKS + ks) * 32 + lane) * 16), acc1);
  }

#pragma unroll
  for (int nt = 0; nt < 2; ++nt) {
    v8f acc = nt ? acc1 : acc0;
    int oc = nt * 16 + (lane & 15);
    float bb = bias[oc];
    float lsum = 0.f, lsq = 0.f;
#pragma unroll
    for (int r = 0; r < 8; ++r) {
      int y = ybase + r + kb8;
      float v = acc[r] + bb;
      out[(((size_t)b * ENC + oc) * T_ + t) * HW + (size_t)xr * W_ + y] = v;
      lsum += v; lsq += v * v;
    }
    atomicAdd(&bsum[oc], lsum);
    atomicAdd(&bsq[oc], lsq);
  }
  __syncthreads();
  if (tid < ENC) {
    atomicAdd(&stats[tid], bsum[tid]);
    atomicAdd(&stats[32 + tid], bsq[tid]);
  }
}

// ---------------------------------------------------------------------------
extern "C" void kernel_launch(void* const* d_in, const int* in_sizes, int n_in,
                              void* d_out, int out_size, void* d_ws, size_t ws_size,
                              hipStream_t stream) {
  const float* x1     = (const float*)d_in[0];
  const float* x2     = (const float*)d_in[1];
  const float* attn   = (const float*)d_in[2];
  const float* W_up   = (const float*)d_in[3];
  const float* b_up   = (const float*)d_in[4];
  const float* W_mlp  = (const float*)d_in[5];
  const float* b_mlp  = (const float*)d_in[6];
  const float* g_mlp  = (const float*)d_in[7];
  const float* be_mlp = (const float*)d_in[8];
  const float* W_c1   = (const float*)d_in[9];
  const float* b_c1   = (const float*)d_in[10];
  const float* g_c1   = (const float*)d_in[11];
  const float* be_c1  = (const float*)d_in[12];
  const float* W_c2   = (const float*)d_in[13];
  const float* b_c2   = (const float*)d_in[14];
  const float* g_c2   = (const float*)d_in[15];
  const float* be_c2  = (const float*)d_in[16];

  char* ws = (char*)d_ws;
  float*  stats  = (float*)(ws + WS_STATS);
  float*  affine = (float*)(ws + WS_AFFINE);
  __bf16* pk_up  = (__bf16*)(ws + WS_PK_UP);
  __bf16* pk_mlp = (__bf16*)(ws + WS_PK_MLP);
  __bf16* pk_c1  = (__bf16*)(ws + WS_PK_C1);
  __bf16* pk_c2  = (__bf16*)(ws + WS_PK_C2);
  float*  hmlp   = (float*)(ws + WS_HMLP);
  __bf16* cat    = (__bf16*)(ws + WS_CAT);
  float*  y1     = (float*)(ws + WS_Y1);
  __bf16* cat2   = (__bf16*)(ws + WS_CAT2);
  float*  y2     = (float*)(ws + WS_Y2);

  // K0: weight repack + zero stats
  k_pack<<<dim3(224), 256, 0, stream>>>(W_up, W_mlp, W_c1, W_c2,
                                        pk_up, pk_mlp, pk_c1, pk_c2, stats);
  // K1: upsample -> cat channels 32..95
  k_upsample<<<dim3(32, 16), 256, 0, stream>>>(x1, b_up, pk_up, cat);
  // K2: attention ctx + MLP -> hmlp (pre-BN) + stats stage 0
  k_attn_mlp<<<dim3(128, 16), 256, 0, stream>>>(x2, attn, b_mlp, pk_mlp, hmlp, stats);
  k_affine<<<1, 32, 0, stream>>>(stats, g_mlp, be_mlp, affine);
  // K3: BN+ReLU hmlp -> cat channels 0..31 (bf16)
  k_bnrelu_bf16<<<dim3(32768), 256, 0, stream>>>(hmlp, affine, cat, CCAT, 0);
  // K4: conv1 96->32 + stats stage 1
  k_conv3x3<CCAT, 27><<<dim3(128, 16), 256, 0, stream>>>(cat, pk_c1, b_c1, y1, stats + 64);
  k_affine<<<1, 32, 0, stream>>>(stats + 64, g_c1, be_c1, affine + 64);
  // K5: BN+ReLU y1 -> cat2 (bf16)
  k_bnrelu_bf16<<<dim3(32768), 256, 0, stream>>>(y1, affine + 64, cat2, ENC, 0);
  // K6: conv2 32->32 + stats stage 2
  k_conv3x3<ENC, 9><<<dim3(128, 16), 256, 0, stream>>>(cat2, pk_c2, b_c2, y2, stats + 128);
  k_affine<<<1, 32, 0, stream>>>(stats + 128, g_c2, be_c2, affine + 128);
  // K7: BN+ReLU y2 -> output (f32)
  k_bnrelu_f32<<<dim3(32768), 256, 0, stream>>>(y2, affine + 128, (float*)d_out);
}
